// ConformerBlock_75110388072855
// MI455X (gfx1250) — compile-verified, matrix operand order checked by reading, not run on hardware
//
#include <hip/hip_runtime.h>

typedef __attribute__((ext_vector_type(16))) __bf16 v16bf;
typedef __attribute__((ext_vector_type(8)))  float  v8f;
typedef unsigned int   u32;
typedef unsigned short u16;

// ---------------------------------------------------------------- utilities
__device__ inline u16 f2bf(float f) {            // fp32 -> bf16 RNE
  u32 u = __builtin_bit_cast(u32, f);
  u32 r = u + 0x7FFFu + ((u >> 16) & 1u);
  return (u16)(r >> 16);
}
__device__ inline float bf2f(u16 h) {
  u32 u = ((u32)h) << 16;
  return __builtin_bit_cast(float, u);
}

struct Frag { u32 d[8]; };                        // 16 packed bf16 = 8 VGPRs
__device__ inline v16bf frag_cast(const Frag& f) {
  return __builtin_bit_cast(v16bf, f);
}
__device__ inline v8f vzero() {
  v8f v;
#pragma unroll
  for (int i = 0; i < 8; ++i) v[i] = 0.f;
  return v;
}
// LDS byte offset of a __shared__ object = low 32 bits of its generic address
__device__ inline u32 lds_off(const void* p) { return (u32)(size_t)p; }

// async global->LDS: 16B per lane, IOFFSET applies to BOTH addresses
__device__ inline void async_b128(u32 lds, const void* gaddr) {
  asm volatile("global_load_async_to_lds_b128 %0, %1, off"
               :: "v"(lds), "v"(gaddr) : "memory");
}
__device__ inline void async_b128_off16(u32 lds, const void* gaddr) {
  asm volatile("global_load_async_to_lds_b128 %0, %1, off offset:16"
               :: "v"(lds), "v"(gaddr) : "memory");
}
__device__ inline void async_b128_off(u32 lds, const void* gaddr, int imm) {
  // imm must be a compile-time constant at each call site (unrolled)
  switch (imm) {
    case 0:  asm volatile("global_load_async_to_lds_b128 %0, %1, off"
                          :: "v"(lds), "v"(gaddr) : "memory"); break;
    case 16: asm volatile("global_load_async_to_lds_b128 %0, %1, off offset:16"
                          :: "v"(lds), "v"(gaddr) : "memory"); break;
    case 32: asm volatile("global_load_async_to_lds_b128 %0, %1, off offset:32"
                          :: "v"(lds), "v"(gaddr) : "memory"); break;
    default: asm volatile("global_load_async_to_lds_b128 %0, %1, off offset:48"
                          :: "v"(lds), "v"(gaddr) : "memory"); break;
  }
}
__device__ inline void wait_async0() {
  asm volatile("s_wait_asynccnt 0x0" ::: "memory");
}

// load A/B fragment from LDS tile stored as u32 rows (2 bf16 per u32).
// 16-bit WMMA layout: lanes 0-15 -> K {0..7,16..23}, lanes 16-31 -> K {8..15,24..31}
__device__ inline void load_frag(Frag& f, const u32* row_base, int hi) {
  const u32* p = row_base + hi * 4;
  *(uint4*)&f.d[0] = *(const uint4*)(p);
  *(uint4*)&f.d[4] = *(const uint4*)(p + 8);
}

// ---------------------------------------------------------------- fp32 -> bf16
__global__ __launch_bounds__(256)
void f32_to_bf16_kernel(const float* __restrict__ src, u16* __restrict__ dst, int n) {
  int i = blockIdx.x * 256 + threadIdx.x;
  if (i < n) dst[i] = f2bf(src[i]);
}

// ---------------------------------------------------------------- LayerNorm
template<bool OUTBF>
__global__ __launch_bounds__(256)
void layernorm_kernel(const float* __restrict__ x, const float* __restrict__ g,
                      const float* __restrict__ bta, u16* __restrict__ outB,
                      float* __restrict__ outF) {
  const int tid = threadIdx.x, wid = tid >> 5, lane = tid & 31;
  const size_t row = (size_t)blockIdx.x * 8 + wid;
  const float* xr = x + row * 512;
  float4 v[4];
  float s = 0.f, sq = 0.f;
#pragma unroll
  for (int c = 0; c < 4; ++c) {
    v[c] = ((const float4*)xr)[c * 32 + lane];
    s  += v[c].x + v[c].y + v[c].z + v[c].w;
    sq += v[c].x * v[c].x + v[c].y * v[c].y + v[c].z * v[c].z + v[c].w * v[c].w;
  }
#pragma unroll
  for (int off = 16; off >= 1; off >>= 1) {
    s  += __shfl_xor(s,  off, 32);
    sq += __shfl_xor(sq, off, 32);
  }
  const float mean = s * (1.f / 512.f);
  const float var  = sq * (1.f / 512.f) - mean * mean;
  const float rstd = rsqrtf(var + 1e-5f);
#pragma unroll
  for (int c = 0; c < 4; ++c) {
    float4 gg = ((const float4*)g)[c * 32 + lane];
    float4 bb = ((const float4*)bta)[c * 32 + lane];
    float y0 = (v[c].x - mean) * rstd * gg.x + bb.x;
    float y1 = (v[c].y - mean) * rstd * gg.y + bb.y;
    float y2 = (v[c].z - mean) * rstd * gg.z + bb.z;
    float y3 = (v[c].w - mean) * rstd * gg.w + bb.w;
    if constexpr (OUTBF) {
      u32 p0 = (u32)f2bf(y0) | ((u32)f2bf(y1) << 16);
      u32 p1 = (u32)f2bf(y2) | ((u32)f2bf(y3) << 16);
      ((uint2*)(outB + row * 512))[c * 32 + lane] = make_uint2(p0, p1);
    } else {
      float4 o; o.x = y0; o.y = y1; o.z = y2; o.w = y3;
      ((float4*)(outF + row * 512))[c * 32 + lane] = o;
    }
  }
}

// ---------------------------------------------------------------- WMMA GEMM
// C[m,n] = sum_k A[m,k] * W[n,k]   (A: MxK bf16 row-major, W: NxK bf16 row-major)
// Double-buffered async global->LDS staging (ASYNCcnt pipeline).
// EPI 0: outB = bf16(acc + biasA)                                   (QKV)
// EPI 1: dual GEMM, outB = bf16(silu(a)*c)                          (FF GLU)
// EPI 2: dual GEMM, outB = bf16(a*sigmoid(c))                       (conv GLU)
// EPI 3: outF = resid + scale*(acc + biasA)                         (out-projs)
template<int EPI>
__global__ __launch_bounds__(256)
void wmma_gemm_kernel(const u16* __restrict__ A, const u16* __restrict__ Wa,
                      const u16* __restrict__ Wc, const float* __restrict__ biasA,
                      const float* __restrict__ biasC, const float* __restrict__ resid,
                      float scale, float* __restrict__ outF, u16* __restrict__ outB,
                      int M, int N, int K, int ldo) {
  constexpr int LDT = 20;                       // u32 stride (16B-aligned rows)
  __shared__ u32 As[2][128 * LDT];
  __shared__ u32 Bs[2][128 * LDT];
  const int tid = threadIdx.x;
  const int wid = tid >> 5, lane = tid & 31;
  const int hi = lane >> 4, ln16 = lane & 15;
  const int wm = (wid & 1) * 64, wn = (wid >> 1) * 32;
  const int m0 = blockIdx.y * 128, n0 = blockIdx.x * 128;
  const int lrow = tid >> 1, lhalf = tid & 1;   // 128 rows x two 16-u16 halves

  constexpr int NPASS = (EPI == 1 || EPI == 2) ? 2 : 1;
  v8f acc[NPASS][4][2];
#pragma unroll
  for (int p = 0; p < NPASS; ++p)
#pragma unroll
    for (int i = 0; i < 4; ++i)
#pragma unroll
      for (int j = 0; j < 2; ++j) acc[p][i][j] = vzero();

  const int nk = K >> 5;
#pragma unroll
  for (int p = 0; p < NPASS; ++p) {
    const u16* W = (p == 0) ? Wa : Wc;
    // issue tile 0 into buffer 0
    {
      u32 la = lds_off(&As[0][lrow * LDT + lhalf * 8]);
      const u16* ga = A + (size_t)(m0 + lrow) * K + lhalf * 16;
      async_b128(la, ga); async_b128_off16(la, ga);
      u32 lb = lds_off(&Bs[0][lrow * LDT + lhalf * 8]);
      const u16* gb = W + (size_t)(n0 + lrow) * K + lhalf * 16;
      async_b128(lb, gb); async_b128_off16(lb, gb);
    }
    int buf = 0;
    for (int kk = 0; kk < nk; ++kk) {
      wait_async0();
      __syncthreads();
      if (kk + 1 < nk) {                         // prefetch next K tile
        const int k0 = (kk + 1) << 5;
        u32 la = lds_off(&As[buf ^ 1][lrow * LDT + lhalf * 8]);
        const u16* ga = A + (size_t)(m0 + lrow) * K + k0 + lhalf * 16;
        async_b128(la, ga); async_b128_off16(la, ga);
        u32 lb = lds_off(&Bs[buf ^ 1][lrow * LDT + lhalf * 8]);
        const u16* gb = W + (size_t)(n0 + lrow) * K + k0 + lhalf * 16;
        async_b128(lb, gb); async_b128_off16(lb, gb);
      }
      Frag af[4], bfr[2];
#pragma unroll
      for (int i = 0; i < 4; ++i)
        load_frag(af[i], &As[buf][(wm + i * 16 + ln16) * LDT], hi);
#pragma unroll
      for (int j = 0; j < 2; ++j)
        load_frag(bfr[j], &Bs[buf][(wn + j * 16 + ln16) * LDT], hi);
#pragma unroll
      for (int i = 0; i < 4; ++i)
#pragma unroll
        for (int j = 0; j < 2; ++j)
          acc[p][i][j] = __builtin_amdgcn_wmma_f32_16x16x32_bf16(
              false, frag_cast(af[i]), false, frag_cast(bfr[j]),
              (short)0, acc[p][i][j], false, false);
      buf ^= 1;
    }
  }

#pragma unroll
  for (int i = 0; i < 4; ++i)
#pragma unroll
    for (int j = 0; j < 2; ++j)
#pragma unroll
      for (int e = 0; e < 8; ++e) {
        const int m = m0 + wm + i * 16 + e + 8 * hi;
        const int n = n0 + wn + j * 16 + ln16;
        float a = acc[0][i][j][e] + biasA[n];
        if constexpr (EPI == 0) {
          outB[(size_t)m * ldo + n] = f2bf(a);
        } else if constexpr (EPI == 1) {
          float c = acc[1][i][j][e] + biasC[n];
          float s = a / (1.f + __expf(-a));              // silu(a) * c
          outB[(size_t)m * ldo + n] = f2bf(s * c);
        } else if constexpr (EPI == 2) {
          float c = acc[1][i][j][e] + biasC[n];
          float sg = 1.f / (1.f + __expf(-c));           // a * sigmoid(c)
          outB[(size_t)m * ldo + n] = f2bf(a * sg);
        } else {
          outF[(size_t)m * ldo + n] = resid[(size_t)m * ldo + n] + scale * a;
        }
      }
}

// ---------------------------------------------------------------- Flash attention
// qkv: [B,S,1536] bf16, channel = t*512 + h*64 + d ; out: [B,S,512] bf16
// grid (B*H, S/128); block 256 (8 waves x 16 query rows). D = 64, key tile 64.
__global__ __launch_bounds__(256)
void flash_attn_kernel(const u16* __restrict__ qkv, u16* __restrict__ out) {
  constexpr int SQ = 36;                        // u32 stride
  __shared__ u32 Qs[128 * SQ];
  __shared__ u32 Ks[64 * SQ];
  __shared__ u32 Vs[64 * SQ];                   // transposed: row=d, packed keys
  __shared__ u16 Ps[8 * 16 * 72];               // per-wave 16x64 P tile
  const int tid = threadIdx.x, wid = tid >> 5, lane = tid & 31;
  const int hi = lane >> 4, ln16 = lane & 15;
  const int b = blockIdx.x >> 3, h = blockIdx.x & 7;
  const int q0 = blockIdx.y * 128;
  const size_t rs = 1536;
  const u16* base = qkv + (size_t)b * 1024 * rs + h * 64;

  { // stage Q tile (128 x 64) asynchronously
    int r = tid >> 1, hf = tid & 1;
    u32 lq = lds_off(Qs + r * SQ + hf * 16);
    const u16* gq = base + (size_t)(q0 + r) * rs + hf * 32;
    async_b128_off(lq, gq, 0);  async_b128_off(lq, gq, 16);
    async_b128_off(lq, gq, 32); async_b128_off(lq, gq, 48);
  }

  float m_run[8], l_run[8];
  v8f O[4];
#pragma unroll
  for (int e = 0; e < 8; ++e) { m_run[e] = -3.0e38f; l_run[e] = 0.f; }
#pragma unroll
  for (int dt = 0; dt < 4; ++dt) O[dt] = vzero();

  for (int kt = 0; kt < 16; ++kt) {
    __syncthreads();
    { // stage K tile (64 x 64) asynchronously, same layout as Q
      int r = tid >> 2, qq = tid & 3;
      u32 lk = lds_off(Ks + r * SQ + qq * 8);
      const u16* gk = base + (size_t)(kt * 64 + r) * rs + 512 + qq * 16;
      async_b128(lk, gk); async_b128_off16(lk, gk);
    }
    { // stage V tile transposed (element-wise, via VGPRs): Vs[d][sk]
      int sk = tid & 63, dg = tid >> 6;
      const u16* src = base + (size_t)(kt * 64 + sk) * rs + 1024 + dg * 16;
      uint4 x0 = ((const uint4*)src)[0], x1 = ((const uint4*)src)[1];
      const u16* dv0 = (const u16*)&x0;
      const u16* dv1 = (const u16*)&x1;
      u16* vs = (u16*)Vs;
#pragma unroll
      for (int i = 0; i < 8; ++i) vs[(dg * 16 + i) * 72 + sk] = dv0[i];
#pragma unroll
      for (int i = 0; i < 8; ++i) vs[(dg * 16 + 8 + i) * 72 + sk] = dv1[i];
    }
    wait_async0();                               // Q (first iter) + K complete
    __syncthreads();

    // S = Q K^T  (per wave: 16 q-rows x 64 keys)
    v8f S[4];
#pragma unroll
    for (int nt = 0; nt < 4; ++nt) S[nt] = vzero();
#pragma unroll
    for (int kk = 0; kk < 64; kk += 32) {
      Frag aq;
      load_frag(aq, Qs + (wid * 16 + ln16) * SQ + kk / 2, hi);
#pragma unroll
      for (int nt = 0; nt < 4; ++nt) {
        Frag bk;
        load_frag(bk, Ks + (nt * 16 + ln16) * SQ + kk / 2, hi);
        S[nt] = __builtin_amdgcn_wmma_f32_16x16x32_bf16(
            false, frag_cast(aq), false, frag_cast(bk), (short)0, S[nt], false, false);
      }
    }
#pragma unroll
    for (int nt = 0; nt < 4; ++nt)
#pragma unroll
      for (int e = 0; e < 8; ++e) S[nt][e] *= 0.125f;   // 1/sqrt(64)

    // online softmax; row r = e + 8*hi lives across 16 lanes of one half-wave
#pragma unroll
    for (int e = 0; e < 8; ++e) {
      float mx = fmaxf(fmaxf(S[0][e], S[1][e]), fmaxf(S[2][e], S[3][e]));
#pragma unroll
      for (int off = 8; off >= 1; off >>= 1) mx = fmaxf(mx, __shfl_xor(mx, off, 32));
      float mn = fmaxf(m_run[e], mx);
      float alpha = __expf(m_run[e] - mn);
      m_run[e] = mn;
      float rsum = 0.f;
#pragma unroll
      for (int nt = 0; nt < 4; ++nt) {
        float p = __expf(S[nt][e] - mn);
        S[nt][e] = p;
        rsum += p;
      }
#pragma unroll
      for (int off = 8; off >= 1; off >>= 1) rsum += __shfl_xor(rsum, off, 32);
      l_run[e] = l_run[e] * alpha + rsum;
#pragma unroll
      for (int dt = 0; dt < 4; ++dt) O[dt][e] *= alpha;
    }

    // round-trip P through wave-private LDS to get A-fragment layout
#pragma unroll
    for (int nt = 0; nt < 4; ++nt)
#pragma unroll
      for (int e = 0; e < 8; ++e)
        Ps[(wid * 16 + e + 8 * hi) * 72 + nt * 16 + ln16] = f2bf(S[nt][e]);
    __asm__ volatile("s_wait_dscnt 0" ::: "memory");

    // O += P V
#pragma unroll
    for (int kk = 0; kk < 64; kk += 32) {
      Frag ap;
      load_frag(ap, (const u32*)(Ps + (wid * 16 + ln16) * 72) + kk / 2, hi);
#pragma unroll
      for (int dt = 0; dt < 4; ++dt) {
        Frag bv;
        load_frag(bv, Vs + (dt * 16 + ln16) * SQ + kk / 2, hi);
        O[dt] = __builtin_amdgcn_wmma_f32_16x16x32_bf16(
            false, frag_cast(ap), false, frag_cast(bv), (short)0, O[dt], false, false);
      }
    }
  }

#pragma unroll
  for (int e = 0; e < 8; ++e) {
    float inv = 1.f / l_run[e];
    int q = q0 + wid * 16 + e + 8 * hi;
    size_t ob = ((size_t)b * 1024 + q) * 512 + h * 64;
#pragma unroll
    for (int dt = 0; dt < 4; ++dt)
      out[ob + dt * 16 + ln16] = f2bf(O[dt][e] * inv);
  }
}

// ---------------------------------------------------------------- grouped dw-conv + BN + SiLU
// in: [B*S, 1024] bf16 (channel-last), groups of 8 channels, kernel 31, pad 15
__global__ __launch_bounds__(256)
void dwconv_bn_silu_kernel(const u16* __restrict__ in, const float* __restrict__ w,
                           const float* __restrict__ bias,
                           const float* __restrict__ bn_g, const float* __restrict__ bn_b,
                           const float* __restrict__ bn_m, const float* __restrict__ bn_v,
                           u16* __restrict__ out) {
  const size_t idx = (size_t)blockIdx.x * 256 + threadIdx.x;
  const int co = (int)(idx & 1023);
  const size_t bs = idx >> 10;
  const int b = (int)(bs >> 10), s = (int)(bs & 1023);
  const int gr = co >> 3;
  const float* wr = w + (size_t)co * 248;      // [8][31]
  float acc = 0.f;
  for (int t = 0; t < 31; ++t) {
    int ss = s + t - 15;
    if ((unsigned)ss < 1024u) {
      const u16* p = in + ((size_t)b * 1024 + ss) * 1024 + gr * 8;
      uint4 d = *(const uint4*)p;
      const u16* dv = (const u16*)&d;
#pragma unroll
      for (int ci = 0; ci < 8; ++ci) acc += bf2f(dv[ci]) * wr[ci * 31 + t];
    }
  }
  float y = acc + bias[co];
  y = (y - bn_m[co]) * rsqrtf(bn_v[co] + 1e-5f) * bn_g[co] + bn_b[co];
  y = y / (1.f + __expf(-y));                  // silu
  out[bs * 1024 + co] = f2bf(y);
}

// ---------------------------------------------------------------- launch
extern "C" void kernel_launch(void* const* d_in, const int* in_sizes, int n_in,
                              void* d_out, int out_size, void* d_ws, size_t ws_size,
                              hipStream_t stream) {
  const int M = 8192;                          // B*S = 8*1024
  const float* x0       = (const float*)d_in[0];
  const float* ff1_g    = (const float*)d_in[1];
  const float* ff1_b    = (const float*)d_in[2];
  const float* ff1_win  = (const float*)d_in[3];
  const float* ff1_bin  = (const float*)d_in[4];
  const float* ff1_wout = (const float*)d_in[5];
  const float* ff1_bout = (const float*)d_in[6];
  const float* attn_g   = (const float*)d_in[7];
  const float* attn_b   = (const float*)d_in[8];
  const float* wqkv     = (const float*)d_in[9];
  const float* bqkv     = (const float*)d_in[10];
  const float* wo       = (const float*)d_in[11];
  const float* bo       = (const float*)d_in[12];
  const float* conv_g   = (const float*)d_in[13];
  const float* conv_b   = (const float*)d_in[14];
  const float* pw1_w    = (const float*)d_in[15];
  const float* pw1_b    = (const float*)d_in[16];
  const float* dw_w     = (const float*)d_in[17];
  const float* dw_b     = (const float*)d_in[18];
  const float* bn_g     = (const float*)d_in[19];
  const float* bn_b     = (const float*)d_in[20];
  const float* bn_m     = (const float*)d_in[21];
  const float* bn_v     = (const float*)d_in[22];
  const float* pw2_w    = (const float*)d_in[23];
  const float* pw2_b    = (const float*)d_in[24];
  const float* ff2_g    = (const float*)d_in[25];
  const float* ff2_b    = (const float*)d_in[26];
  const float* ff2_win  = (const float*)d_in[27];
  const float* ff2_bin  = (const float*)d_in[28];
  const float* ff2_wout = (const float*)d_in[29];
  const float* ff2_bout = (const float*)d_in[30];
  const float* fn_g     = (const float*)d_in[31];
  const float* fn_b     = (const float*)d_in[32];

  size_t off = 0;
  auto carve = [&](size_t bytes) -> void* {
    void* p = (char*)d_ws + off;
    off += (bytes + 255) & ~(size_t)255;
    return p;
  };
  float* xa    = (float*)carve((size_t)M * 512 * 4);   // running residual stream
  u16* h16     = (u16*)carve((size_t)M * 512 * 2);     // LN output (bf16)
  u16* g16     = (u16*)carve((size_t)M * 2048 * 2);    // GLU outputs
  u16* qkv16   = (u16*)carve((size_t)M * 1536 * 2);
  u16* ao16    = (u16*)carve((size_t)M * 512 * 2);
  u16* c16     = (u16*)carve((size_t)M * 1024 * 2);
  u16* w_ff1i  = (u16*)carve((size_t)4096 * 512 * 2);
  u16* w_ff1o  = (u16*)carve((size_t)512 * 2048 * 2);
  u16* w_qkv   = (u16*)carve((size_t)1536 * 512 * 2);
  u16* w_wo    = (u16*)carve((size_t)512 * 512 * 2);
  u16* w_pw1   = (u16*)carve((size_t)2048 * 512 * 2);
  u16* w_pw2   = (u16*)carve((size_t)512 * 1024 * 2);
  u16* w_ff2i  = (u16*)carve((size_t)4096 * 512 * 2);
  u16* w_ff2o  = (u16*)carve((size_t)512 * 2048 * 2);

  auto cvt = [&](const float* s, u16* d, int n) {
    f32_to_bf16_kernel<<<(n + 255) / 256, 256, 0, stream>>>(s, d, n);
  };
  cvt(ff1_win,  w_ff1i, 4096 * 512);
  cvt(ff1_wout, w_ff1o, 512 * 2048);
  cvt(wqkv,     w_qkv,  1536 * 512);
  cvt(wo,       w_wo,   512 * 512);
  cvt(pw1_w,    w_pw1,  2048 * 512);
  cvt(pw2_w,    w_pw2,  512 * 1024);
  cvt(ff2_win,  w_ff2i, 4096 * 512);
  cvt(ff2_wout, w_ff2o, 512 * 2048);

  const dim3 blk(256);
  const dim3 gLN(M / 8);

  // ---- FF1 (half-step): xa = x0 + 0.5*(silu(a)*c @ Wout + bout)
  layernorm_kernel<true><<<gLN, blk, 0, stream>>>(x0, ff1_g, ff1_b, h16, nullptr);
  wmma_gemm_kernel<1><<<dim3(2048 / 128, M / 128), blk, 0, stream>>>(
      h16, w_ff1i, w_ff1i + (size_t)2048 * 512, ff1_bin, ff1_bin + 2048,
      nullptr, 0.f, nullptr, g16, M, 2048, 512, 2048);
  wmma_gemm_kernel<3><<<dim3(512 / 128, M / 128), blk, 0, stream>>>(
      g16, w_ff1o, nullptr, ff1_bout, nullptr, x0, 0.5f, xa, nullptr, M, 512, 2048, 512);

  // ---- MHSA: xa += proj(flash_attn(qkv(LN(xa))))
  layernorm_kernel<true><<<gLN, blk, 0, stream>>>(xa, attn_g, attn_b, h16, nullptr);
  wmma_gemm_kernel<0><<<dim3(1536 / 128, M / 128), blk, 0, stream>>>(
      h16, w_qkv, nullptr, bqkv, nullptr, nullptr, 0.f, nullptr, qkv16, M, 1536, 512, 1536);
  flash_attn_kernel<<<dim3(64, 8), blk, 0, stream>>>(qkv16, ao16);
  wmma_gemm_kernel<3><<<dim3(512 / 128, M / 128), blk, 0, stream>>>(
      ao16, w_wo, nullptr, bo, nullptr, xa, 1.0f, xa, nullptr, M, 512, 512, 512);

  // ---- Conv module: xa += pw2(silu(BN(dwconv(GLU(pw1(LN(xa)))))))
  layernorm_kernel<true><<<gLN, blk, 0, stream>>>(xa, conv_g, conv_b, h16, nullptr);
  wmma_gemm_kernel<2><<<dim3(1024 / 128, M / 128), blk, 0, stream>>>(
      h16, w_pw1, w_pw1 + (size_t)1024 * 512, pw1_b, pw1_b + 1024,
      nullptr, 0.f, nullptr, g16, M, 1024, 512, 1024);
  dwconv_bn_silu_kernel<<<(size_t)M * 1024 / 256, blk, 0, stream>>>(
      g16, dw_w, dw_b, bn_g, bn_b, bn_m, bn_v, c16);
  wmma_gemm_kernel<3><<<dim3(512 / 128, M / 128), blk, 0, stream>>>(
      c16, w_pw2, nullptr, pw2_b, nullptr, xa, 1.0f, xa, nullptr, M, 512, 1024, 512);

  // ---- FF2 (half-step)
  layernorm_kernel<true><<<gLN, blk, 0, stream>>>(xa, ff2_g, ff2_b, h16, nullptr);
  wmma_gemm_kernel<1><<<dim3(2048 / 128, M / 128), blk, 0, stream>>>(
      h16, w_ff2i, w_ff2i + (size_t)2048 * 512, ff2_bin, ff2_bin + 2048,
      nullptr, 0.f, nullptr, g16, M, 2048, 512, 2048);
  wmma_gemm_kernel<3><<<dim3(512 / 128, M / 128), blk, 0, stream>>>(
      g16, w_ff2o, nullptr, ff2_bout, nullptr, xa, 0.5f, xa, nullptr, M, 512, 2048, 512);

  // ---- final LayerNorm -> d_out (fp32)
  layernorm_kernel<false><<<gLN, blk, 0, stream>>>(xa, fn_g, fn_b, nullptr, (float*)d_out);
}
